// PDUSwinNet_9543417332056
// MI455X (gfx1250) — compile-verified, matrix operand order checked by reading, not run on hardware
//
#include <hip/hip_runtime.h>
#include <math.h>

// ---------------- CDNA5 WMMA types ----------------
typedef _Float16 v16h __attribute__((ext_vector_type(16)));
typedef _Float16 v8h  __attribute__((ext_vector_type(8)));
typedef float    v8f  __attribute__((ext_vector_type(8)));

#define WS_     7
#define N_TOK   49
#define C_DIM   256
#define NH_     8
#define HD_     32
#define SHIFT_  3
#define NWIN    64              // 8x8 windows per image
#define BATCH   16
#define L_TOK   3136            // 56*56
#define M_ROWS  (BATCH * L_TOK) // 50176
#define SCALE_  0.17677669529663687f // 1/sqrt(32)

// A-style fragment load (also used for B via B^T): lane = row (0..15),
// per ISA 7.12.2: lanes<16 hold K in {koff..koff+7, koff+16..koff+23}, koff=0;
// lanes>=16 same pattern with koff=8.  q must point at (row,k0+koff).
__device__ __forceinline__ v16h ld_frag(const _Float16* q) {
    v16h f;
#pragma unroll
    for (int e = 0; e < 8; ++e) {
        f[e]     = q[e];
        f[e + 8] = q[e + 16];
    }
    return f;
}

__device__ __forceinline__ v8f wmma16(v16h a, v16h b, v8f c) {
    return __builtin_amdgcn_wmma_f32_16x16x32_f16(false, a, false, b,
                                                  (short)0, c, false, false);
}

__device__ __forceinline__ int region3(int x) { // shifted-window region id
    return (x < 49) ? 0 : ((x < 53) ? 1 : 2);
}

// windowed row -> source/target token index (roll by `shift` folded in)
__device__ __forceinline__ int win_row_to_token(int m, int shift) {
    const int win = m / N_TOK, t = m - win * N_TOK;
    const int b = win >> 6, wi = win & 63;
    const int wh = wi >> 3, ww = wi & 7;
    int gh = wh * 7 + t / 7;
    int gw = ww * 7 + t % 7;
    gh += shift; if (gh >= 56) gh -= 56;
    gw += shift; if (gw >= 56) gw -= 56;
    return b * L_TOK + gh * 56 + gw;
}

// ---------------- f32 -> f16 weight convert ----------------
__global__ void cvt_k(const float* __restrict__ in, _Float16* __restrict__ out, int n) {
    int i = blockIdx.x * 256 + threadIdx.x;
    if (i < n) out[i] = (_Float16)in[i];
}

// ---------------- patch embed (4x4/4 conv) + LayerNorm ----------------
__global__ __launch_bounds__(256) void patch_embed_k(
    const float* __restrict__ x, const float* __restrict__ pw,
    const float* __restrict__ pb, const float* __restrict__ ng,
    const float* __restrict__ nb, float* __restrict__ X) {
    __shared__ float px[16];
    __shared__ float red[256];
    const int tok = blockIdx.x;
    const int b = tok / L_TOK, t = tok - b * L_TOK;
    const int gh = t / 56, gw = t - gh * 56;
    const int tid = threadIdx.x;
    if (tid < 16) {
        const int i = tid >> 2, j = tid & 3;
        px[tid] = x[(size_t)b * 224 * 224 + (gh * 4 + i) * 224 + gw * 4 + j];
    }
    __syncthreads();
    float s = pb[tid];
#pragma unroll
    for (int p = 0; p < 16; ++p) s = fmaf(px[p], pw[tid * 16 + p], s);
    // LayerNorm over C=256
    red[tid] = s; __syncthreads();
    for (int st = 128; st > 0; st >>= 1) { if (tid < st) red[tid] += red[tid + st]; __syncthreads(); }
    const float mean = red[0] * (1.0f / 256.0f);
    __syncthreads();
    const float d = s - mean;
    red[tid] = d * d; __syncthreads();
    for (int st = 128; st > 0; st >>= 1) { if (tid < st) red[tid] += red[tid + st]; __syncthreads(); }
    const float var = red[0] * (1.0f / 256.0f);
    X[(size_t)tok * C_DIM + tid] = d * rsqrtf(var + 1e-5f) * ng[tid] + nb[tid];
}

// ---------------- LayerNorm (+ optional window partition gather), f16 out ----
template <bool PART>
__global__ __launch_bounds__(256) void ln_k(
    const float* __restrict__ X, const float* __restrict__ g,
    const float* __restrict__ b, _Float16* __restrict__ out, int shift) {
    __shared__ float red[256];
    const int r = blockIdx.x, tid = threadIdx.x;
    const int src = PART ? win_row_to_token(r, shift) : r;
    const float v = X[(size_t)src * C_DIM + tid];
    red[tid] = v; __syncthreads();
    for (int st = 128; st > 0; st >>= 1) { if (tid < st) red[tid] += red[tid + st]; __syncthreads(); }
    const float mean = red[0] * (1.0f / 256.0f);
    __syncthreads();
    const float d = v - mean;
    red[tid] = d * d; __syncthreads();
    for (int st = 128; st > 0; st >>= 1) { if (tid < st) red[tid] += red[tid + st]; __syncthreads(); }
    const float var = red[0] * (1.0f / 256.0f);
    out[(size_t)r * C_DIM + tid] = (_Float16)(d * rsqrtf(var + 1e-5f) * g[tid] + b[tid]);
}

// ---------------- tiled WMMA GEMM:  C[M,N] = A[M,K] * W[N,K]^T + epilogue ----
// block 256 thr = 8 waves; block tile 128x128; wave tile 32x64 (2x4 frags).
// Double-buffered LDS staging of 128x32 A/B panels: each thread moves one
// 32-byte half-row per k-step (global_load_b128 -> ds_store_b128), cutting
// L2 read traffic 3x vs direct per-wave fragment loads.
// EPI: 0 = +bias, f16 store (QKV)         1 = +bias, window-reverse residual
//      2 = +bias, GELU, f16 store (fc1)   3 = +bias, residual add (fc2)
template <int EPI>
__global__ __launch_bounds__(256) void gemm_wmma_k(
    const _Float16* __restrict__ A, const _Float16* __restrict__ W,
    const float* __restrict__ bias, _Float16* __restrict__ outH,
    float* __restrict__ X, int N, int K, int shift) {
    __shared__ _Float16 sA[2][128 * 40];
    __shared__ _Float16 sB[2][128 * 40];

    const int tid = threadIdx.x;
    const int wave = tid >> 5, lane = tid & 31;
    const int wm = wave & 3, wn = wave >> 2;
    const int arow = lane & 15;
    const int koff = (lane & 16) ? 8 : 0;

    // staging role: 2 threads per row, 16 halves (32B) each
    const int srow = tid >> 1;
    const int sseg = (tid & 1) * 16;
    const _Float16* gA = A + (size_t)(blockIdx.x * 128 + srow) * K + sseg;
    const _Float16* gB = W + (size_t)(blockIdx.y * 128 + srow) * K + sseg;

    v8f acc[2][4];
#pragma unroll
    for (int i = 0; i < 2; ++i)
#pragma unroll
        for (int j = 0; j < 4; ++j)
            acc[i][j] = (v8f){0.f, 0.f, 0.f, 0.f, 0.f, 0.f, 0.f, 0.f};

    // preload first panel into registers
    v8h ra0 = *(const v8h*)(gA);
    v8h ra1 = *(const v8h*)(gA + 8);
    v8h rb0 = *(const v8h*)(gB);
    v8h rb1 = *(const v8h*)(gB + 8);

    int p = 0;
    for (int k = 0; k < K; k += 32) {
        {   // park staged registers in LDS buffer p
            _Float16* a = &sA[p][srow * 40 + sseg];
            *(v8h*)a = ra0; *(v8h*)(a + 8) = ra1;
            _Float16* b = &sB[p][srow * 40 + sseg];
            *(v8h*)b = rb0; *(v8h*)(b + 8) = rb1;
        }
        __syncthreads();
        if (k + 32 < K) {   // issue next panel's global loads (overlap compute)
            ra0 = *(const v8h*)(gA + k + 32);
            ra1 = *(const v8h*)(gA + k + 40);
            rb0 = *(const v8h*)(gB + k + 32);
            rb1 = *(const v8h*)(gB + k + 40);
            if (k + 64 < K) {   // gfx1250 global_prefetch for the panel after
                __builtin_prefetch(gA + k + 64, 0, 1);
                __builtin_prefetch(gB + k + 64, 0, 1);
            }
        }
        v16h af[2], bf[4];
        af[0] = ld_frag(&sA[p][(wm * 32 + arow) * 40 + koff]);
        af[1] = ld_frag(&sA[p][(wm * 32 + 16 + arow) * 40 + koff]);
#pragma unroll
        for (int j = 0; j < 4; ++j)
            bf[j] = ld_frag(&sB[p][(wn * 64 + j * 16 + arow) * 40 + koff]);
#pragma unroll
        for (int i = 0; i < 2; ++i)
#pragma unroll
            for (int j = 0; j < 4; ++j)
                acc[i][j] = wmma16(af[i], bf[j], acc[i][j]);
        __syncthreads();
        p ^= 1;
    }

    const int ccol = lane & 15;
    const int rb = (lane >> 4) * 8;
    const int row0 = blockIdx.x * 128 + wm * 32;
    const int col0 = blockIdx.y * 128 + wn * 64;
#pragma unroll
    for (int i = 0; i < 2; ++i) {
#pragma unroll
        for (int j = 0; j < 4; ++j) {
            const int ncol = col0 + j * 16 + ccol;
            const float bb = bias[ncol];
#pragma unroll
            for (int e = 0; e < 8; ++e) {
                const int m = row0 + i * 16 + rb + e;
                float v = acc[i][j][e] + bb;
                if (EPI == 0) {
                    outH[(size_t)m * N + ncol] = (_Float16)v;
                } else if (EPI == 1) {
                    const int tok = win_row_to_token(m, shift);
                    const size_t idx = (size_t)tok * C_DIM + ncol;
                    X[idx] = X[idx] + v;           // shortcut + proj(o)
                } else if (EPI == 2) {
                    const float gv = 0.5f * v * (1.0f + erff(v * 0.70710678f));
                    outH[(size_t)m * N + ncol] = (_Float16)gv;
                } else {
                    const size_t idx = (size_t)m * C_DIM + ncol;
                    X[idx] = X[idx] + v;           // x + mlp(x)
                }
            }
        }
    }
}

// ---------------- windowed attention: one block per (window, head) ----------
// 128 thr = 4 waves; wave w owns M-tile w (rows 16w..16w+15 of the padded 64).
__global__ __launch_bounds__(128) void attn_k(
    const _Float16* __restrict__ qkv, const float* __restrict__ rpb,
    _Float16* __restrict__ o, int shift) {
    __shared__ _Float16 sQ[64 * 40];
    __shared__ _Float16 sK[64 * 40];
    __shared__ _Float16 sVt[32 * 72];   // V transposed: [d][tok]
    __shared__ float    sS[64 * 68];
    __shared__ _Float16 sP[64 * 72];

    const int win = blockIdx.x;
    const int head = blockIdx.y;
    const int tid = threadIdx.x;
    const int wi = win & 63;
    const int wh = wi >> 3, ww = wi & 7;

    // stage q/k/v (zero-pad tokens 49..63)
    const _Float16* base = qkv + (size_t)win * N_TOK * 768 + head * HD_;
    for (int idx = tid; idx < 64 * 32; idx += 128) {
        const int t = idx >> 5, d = idx & 31;
        _Float16 qv = (_Float16)0.f, kv = (_Float16)0.f, vv = (_Float16)0.f;
        if (t < N_TOK) {
            const _Float16* row = base + (size_t)t * 768;
            qv = row[d]; kv = row[256 + d]; vv = row[512 + d];
        }
        sQ[t * 40 + d] = qv;
        sK[t * 40 + d] = kv;
        sVt[d * 72 + t] = vv;
    }
    __syncthreads();

    const int wave = tid >> 5, lane = tid & 31;
    const int arow = lane & 15;
    const int koff = (lane & 16) ? 8 : 0;

    // S = Q K^T  (K=32 == HD, single WMMA per 16x16 tile)
    const v16h aq = ld_frag(sQ + (wave * 16 + arow) * 40 + koff);
    v8f sacc[4];
#pragma unroll
    for (int j = 0; j < 4; ++j) {
        const v16h bk = ld_frag(sK + (j * 16 + arow) * 40 + koff);
        v8f z = (v8f){0.f, 0.f, 0.f, 0.f, 0.f, 0.f, 0.f, 0.f};
        sacc[j] = wmma16(aq, bk, z);
    }
    const int ccol = lane & 15, rb = (lane >> 4) * 8;
#pragma unroll
    for (int j = 0; j < 4; ++j)
#pragma unroll
        for (int e = 0; e < 8; ++e)
            sS[(wave * 16 + rb + e) * 68 + j * 16 + ccol] = sacc[j][e];
    __syncthreads();

    // softmax: lane handles row lane&15, columns [32*(lane>>4), +32)
    const int r = lane & 15;
    const int m_tok = wave * 16 + r;
    const int c0 = (lane >> 4) * 32;
    if (m_tok < N_TOK) {
        const int tr = m_tok / 7, tc = m_tok - tr * 7;
        int id_m = 0;
        if (shift) {
            const int gh = wh * 7 + tr, gw = ww * 7 + tc;
            id_m = region3(gh) * 3 + region3(gw);
        }
        float buf[32];
        float mx = -3.0e38f;
#pragma unroll
        for (int cc = 0; cc < 32; ++cc) {
            const int c = c0 + cc;
            float v = -3.0e38f;
            if (c < N_TOK) {
                const int nr = c / 7, nc = c - nr * 7;
                const int ridx = (tr - nr + 6) * 13 + (tc - nc + 6);
                v = sS[m_tok * 68 + c] * SCALE_ + rpb[ridx * NH_ + head];
                if (shift) {
                    const int gh = wh * 7 + nr, gw = ww * 7 + nc;
                    const int id_n = region3(gh) * 3 + region3(gw);
                    if (id_n != id_m) v -= 100.0f;
                }
            }
            buf[cc] = v;
            mx = fmaxf(mx, v);
        }
        mx = fmaxf(mx, __shfl_xor(mx, 16, 32));
        float se = 0.f;
#pragma unroll
        for (int cc = 0; cc < 32; ++cc) {
            const float ex = __expf(buf[cc] - mx);
            buf[cc] = ex;
            se += ex;
        }
        se += __shfl_xor(se, 16, 32);
        const float inv = 1.0f / se;
#pragma unroll
        for (int cc = 0; cc < 32; ++cc)
            sP[m_tok * 72 + c0 + cc] = (_Float16)(buf[cc] * inv);
    } else {
#pragma unroll
        for (int cc = 0; cc < 32; ++cc)
            sP[m_tok * 72 + c0 + cc] = (_Float16)0.f;
    }
    __syncthreads();

    // O = P V : K = 64 (2 k-steps), N = 32 (2 tiles)
    v8f oacc[2];
    oacc[0] = (v8f){0.f, 0.f, 0.f, 0.f, 0.f, 0.f, 0.f, 0.f};
    oacc[1] = (v8f){0.f, 0.f, 0.f, 0.f, 0.f, 0.f, 0.f, 0.f};
#pragma unroll
    for (int kk = 0; kk < 2; ++kk) {
        const v16h ap = ld_frag(sP + (wave * 16 + arow) * 72 + kk * 32 + koff);
#pragma unroll
        for (int j = 0; j < 2; ++j) {
            const v16h bv = ld_frag(sVt + (j * 16 + arow) * 72 + kk * 32 + koff);
            oacc[j] = wmma16(ap, bv, oacc[j]);
        }
    }
#pragma unroll
    for (int j = 0; j < 2; ++j)
#pragma unroll
        for (int e = 0; e < 8; ++e) {
            const int m = wave * 16 + rb + e;
            if (m < N_TOK)
                o[((size_t)win * N_TOK + m) * C_DIM + head * HD_ + j * 16 + ccol] =
                    (_Float16)oacc[j][e];
        }
}

// ---------------- launcher ----------------
extern "C" void kernel_launch(void* const* d_in, const int* in_sizes, int n_in,
                              void* d_out, int out_size, void* d_ws, size_t ws_size,
                              hipStream_t stream) {
    const float* x      = (const float*)d_in[0];
    const float* pe_w   = (const float*)d_in[1];
    const float* pe_b   = (const float*)d_in[2];
    const float* pe_ng  = (const float*)d_in[3];
    const float* pe_nb  = (const float*)d_in[4];
    const float* n1g    = (const float*)d_in[5];
    const float* n1b    = (const float*)d_in[6];
    const float* qkv_w  = (const float*)d_in[7];
    const float* qkv_b  = (const float*)d_in[8];
    const float* rpb    = (const float*)d_in[9];
    const float* proj_w = (const float*)d_in[10];
    const float* proj_b = (const float*)d_in[11];
    const float* n2g    = (const float*)d_in[12];
    const float* n2b    = (const float*)d_in[13];
    const float* w1     = (const float*)d_in[14];
    const float* b1     = (const float*)d_in[15];
    const float* w2     = (const float*)d_in[16];
    const float* b2     = (const float*)d_in[17];

    float* X = (float*)d_out;                        // [M_ROWS, 256] f32, in-place residual
    char* ws = (char*)d_ws;
    _Float16* bufA = (_Float16*)ws;                                      // 50176x1024 f16
    _Float16* bufB = (_Float16*)(ws + (size_t)M_ROWS * 1024 * 2);        // 50176x768  f16
    _Float16* wW   = (_Float16*)(ws + (size_t)M_ROWS * 1024 * 2
                                    + (size_t)M_ROWS * 768 * 2);         // f16 weights

    // weight layout in wW (halves): qkv[2]@0, proj[2]@393216, w1[2]@524288, w2[2]@1048576
    cvt_k<<<(2 * 768 * 256 + 255) / 256, 256, 0, stream>>>(qkv_w, wW, 2 * 768 * 256);
    cvt_k<<<(2 * 256 * 256 + 255) / 256, 256, 0, stream>>>(proj_w, wW + 393216, 2 * 256 * 256);
    cvt_k<<<(2 * 1024 * 256 + 255) / 256, 256, 0, stream>>>(w1, wW + 524288, 2 * 1024 * 256);
    cvt_k<<<(2 * 256 * 1024 + 255) / 256, 256, 0, stream>>>(w2, wW + 1048576, 2 * 256 * 1024);

    patch_embed_k<<<M_ROWS, 256, 0, stream>>>(x, pe_w, pe_b, pe_ng, pe_nb, X);

    for (int i = 0; i < 2; ++i) {
        const int shift = i ? SHIFT_ : 0;
        // LN1 + window partition (with roll) -> bufA f16 [M,256]
        ln_k<true><<<M_ROWS, 256, 0, stream>>>(X, n1g + i * 256, n1b + i * 256, bufA, shift);
        // QKV: [M,256]x[768,256]^T -> bufB f16 [M,768]
        gemm_wmma_k<0><<<dim3(M_ROWS / 128, 6), 256, 0, stream>>>(
            bufA, wW + i * 196608, qkv_b + i * 768, bufB, nullptr, 768, 256, 0);
        // attention per (window, head) -> bufA f16 [M,256] (windowed order)
        attn_k<<<dim3(BATCH * NWIN, NH_), 128, 0, stream>>>(
            bufB, rpb + i * 169 * NH_, bufA, shift);
        // proj + window-reverse (un-roll) + residual into X
        gemm_wmma_k<1><<<dim3(M_ROWS / 128, 2), 256, 0, stream>>>(
            bufA, wW + 393216 + i * 65536, proj_b + i * 256, nullptr, X, 256, 256, shift);
        // LN2 -> bufB f16 [M,256]
        ln_k<false><<<M_ROWS, 256, 0, stream>>>(X, n2g + i * 256, n2b + i * 256, bufB, 0);
        // fc1 + GELU -> bufA f16 [M,1024]
        gemm_wmma_k<2><<<dim3(M_ROWS / 128, 8), 256, 0, stream>>>(
            bufB, wW + 524288 + i * 262144, b1 + i * 1024, bufA, nullptr, 1024, 256, 0);
        // fc2 + residual into X
        gemm_wmma_k<3><<<dim3(M_ROWS / 128, 2), 256, 0, stream>>>(
            bufA, wW + 1048576 + i * 262144, b2 + i * 256, nullptr, X, 256, 1024, 0);
    }
}
